// MultiHeadFlashAttention_78898549227744
// MI455X (gfx1250) — matrix operand order repaired and checked by reading, not compiled, not run on hardware
//
#include <hip/hip_runtime.h>
#include <hip/hip_bf16.h>

// ---------------------------------------------------------------------------
// MHA forward for B=2, S=2048, E=1024, H=16, D=64 on gfx1250 (wave32, WMMA,
// TDM async staging of K/V tiles).
// ---------------------------------------------------------------------------

constexpr int kB = 2;
constexpr int kS = 2048;
constexpr int kE = 1024;
constexpr int kH = 16;
constexpr int kD = 64;
constexpr int kM = kB * kS;        // 4096 rows in the [B*S, E] view

typedef __attribute__((ext_vector_type(16))) _Float16     v16h;
typedef __attribute__((ext_vector_type(8)))  _Float16     v8h;
typedef __attribute__((ext_vector_type(4)))  _Float16     v4h;
typedef __attribute__((ext_vector_type(8)))  float        v8f;
typedef __attribute__((ext_vector_type(4)))  float        v4f;
typedef __attribute__((ext_vector_type(4)))  unsigned int v4u;
typedef __attribute__((ext_vector_type(4)))  int          i4;
typedef __attribute__((ext_vector_type(8)))  int          i8;

union V16H { v16h v; v4u u[2]; };

__device__ __forceinline__ v8f wmma_f16(v16h a, v16h b, v8f c) {
  // D = A(16x32 f16) * B(32x16 f16) + C(16x16 f32)
  return __builtin_amdgcn_wmma_f32_16x16x32_f16(
      /*neg_a=*/false, a, /*neg_b=*/false, b,
      /*c_mod=*/(short)0, c, /*reuse_a=*/false, /*reuse_b=*/false);
}

// ---------------------------------------------------------------------------
// TDM: async 2D-tile load global -> LDS (D# per CDNA5 ISA ch.8).
//   data_size = 8B units; tile = tile_d0_units x tile_rows; row stride =
//   stride_units; LDS row padding = (pad_amt_code+1) DWORDs after every
//   2^(pad_int_code+1) DWORDs (set to one row) for bank-conflict-free reads.
// ---------------------------------------------------------------------------
__device__ __forceinline__ void tdm_load_2d(unsigned lds_off, const void* gaddr,
                                            unsigned tile_d0_units,
                                            unsigned tile_rows,
                                            unsigned stride_units,
                                            unsigned pad_int_code,
                                            unsigned pad_amt_code) {
  const unsigned long long ga = (unsigned long long)(uintptr_t)gaddr;
  // group 0: count=1 | lds_addr[63:32] | global_addr[120:64] | type=2[127:126]
  v4u g0;
  g0[0] = 1u;
  g0[1] = lds_off;
  g0[2] = (unsigned)ga;
  g0[3] = (unsigned)((ga >> 32) & 0x01FFFFFFu) | 0x80000000u;
  // group 1 (256b): wg_mask=0, data_size=3(8B), pad_enable, pad codes,
  // tensor_dim0=stride, tensor_dim1=big, tile_dim0/1/2, tensor_dim0_stride.
  const unsigned long long dim0 = stride_units;
  const unsigned long long dim1 = 1u << 20;
  unsigned long long q0 = (3ull << 16) | (1ull << 20) |
                          ((unsigned long long)pad_int_code << 22) |
                          ((unsigned long long)pad_amt_code << 25) |
                          ((dim0 & 0xFFFFull) << 48);
  unsigned long long q1 = (dim0 >> 16) | (dim1 << 16) |
                          ((unsigned long long)tile_d0_units << 48);
  unsigned long long q2 = (unsigned long long)tile_rows |
                          ((unsigned long long)stride_units << 32);
  unsigned long long q3 = 0ull;  // stride0 high bits (0), dim1_stride unused (2D)
  i8 g1;
  g1[0] = (int)q0; g1[1] = (int)(q0 >> 32);
  g1[2] = (int)q1; g1[3] = (int)(q1 >> 32);
  g1[4] = (int)q2; g1[5] = (int)(q2 >> 32);
  g1[6] = (int)q3; g1[7] = (int)(q3 >> 32);
  const i4 z4 = {0, 0, 0, 0};
  const i8 z8 = {0, 0, 0, 0, 0, 0, 0, 0};
  // clang-23 / therock-10.0 form: 6 args (g0, g1, g2, g3, extra, cpol)
  __builtin_amdgcn_tensor_load_to_lds(g0, g1, z4, z4, z8, 0);
}

// ---------------------------------------------------------------------------
// fp32 -> fp16 conversion, 4-wide grid-stride
// ---------------------------------------------------------------------------
__global__ void cvt_f32_to_f16(const v4f* __restrict__ src,
                               v4h* __restrict__ dst, int n4) {
  int i = blockIdx.x * blockDim.x + threadIdx.x;
  const int stride = gridDim.x * blockDim.x;
  for (; i < n4; i += stride) {
    v4f f = src[i];
    v4h h;
    h[0] = (_Float16)f[0]; h[1] = (_Float16)f[1];
    h[2] = (_Float16)f[2]; h[3] = (_Float16)f[3];
    dst[i] = h;
  }
}

// ---------------------------------------------------------------------------
// GEMM: Y[m,n] = sum_k X[m,k] * W[n,k] + bias[n]
// One wave computes a 16(M) x 64(N) tile: 4 accumulators, K swept in 32s.
// MODE 0: Q -> f16, scaled by 1/sqrt(D); 1: K -> f16; 2: V -> f16 transposed
// [B,H,D,S]; 3: out -> f32 (d_out).
// ---------------------------------------------------------------------------
template <int MODE>
__global__ __launch_bounds__(128)
void gemm_kernel(const _Float16* __restrict__ X, const _Float16* __restrict__ W,
                 const float* __restrict__ bias, _Float16* __restrict__ Yh,
                 float* __restrict__ Yf) {
  const int lane = threadIdx.x & 31;
  const int wv   = threadIdx.x >> 5;
  const int lm   = lane & 15;
  const int hi   = lane >> 4;
  const int m0   = (blockIdx.x * 4 + wv) * 16;
  const int n0   = blockIdx.y * 64;

  const _Float16* arow  = X + (size_t)(m0 + lm) * kE;
  const _Float16* brow0 = W + (size_t)(n0 + lm) * kE;

  v8f c[4] = {{}, {}, {}, {}};
  for (int k = 0; k < kE; k += 32) {
    V16H a;
    a.u[0] = *(const v4u*)(arow + k + hi * 8);
    a.u[1] = *(const v4u*)(arow + k + 16 + hi * 8);
#pragma unroll
    for (int t = 0; t < 4; ++t) {
      const _Float16* br = brow0 + (size_t)t * 16 * kE;
      V16H bm;
      bm.u[0] = *(const v4u*)(br + k + hi * 8);
      bm.u[1] = *(const v4u*)(br + k + 16 + hi * 8);
      c[t] = wmma_f16(a.v, bm.v, c[t]);
    }
  }

#pragma unroll
  for (int t = 0; t < 4; ++t) {
    const int n = n0 + t * 16 + lm;
    const float bb = bias[n];
    if (MODE == 3) {
#pragma unroll
      for (int j = 0; j < 8; ++j)
        Yf[(size_t)(m0 + hi * 8 + j) * kE + n] = c[t][j] + bb;
    } else if (MODE == 2) {
      v8h pk;
#pragma unroll
      for (int j = 0; j < 8; ++j) pk[j] = (_Float16)(c[t][j] + bb);
      const int bidx = m0 / kS;
      const int s    = (m0 % kS) + hi * 8;
      const int hh   = n / kD, dd = n % kD;
      *(v8h*)(Yh + ((size_t)((bidx * kH + hh) * kD + dd)) * kS + s) = pk;
    } else {
      const float sc = (MODE == 0) ? 0.125f : 1.0f;  // fold sm_scale into Q
#pragma unroll
      for (int j = 0; j < 8; ++j)
        Yh[(size_t)(m0 + hi * 8 + j) * kE + n] = (_Float16)((c[t][j] + bb) * sc);
    }
  }
}

// ---------------------------------------------------------------------------
// Flash attention, transposed formulation, TDM-staged K/V tiles.
//   scoresT[k,q] = K(16x32d) x QT(32d x 16q); OT[d,q] = Vt(16d x 32k) x PT.
// Block = 4 waves, all sharing (b,h): K tile (32x128B) and Vt tile (64x64B)
// per iteration are fetched ONCE by the TDM into double-buffered LDS
// (TENSORcnt + barriers), padded 16B/row for conflict-free ds_load_b128.
// ---------------------------------------------------------------------------
__global__ __launch_bounds__(128)
void flash_kernel(const _Float16* __restrict__ Qm, const _Float16* __restrict__ Km,
                  const _Float16* __restrict__ Vt, _Float16* __restrict__ Om) {
  __shared__ __align__(16) _Float16 Kbuf[2][32][72];  // 64+8 halves/row
  __shared__ __align__(16) _Float16 Vbuf[2][64][40];  // 32+8 halves/row
  __shared__ __align__(16) _Float16 plds[4][16][40];

  const int lane = threadIdx.x & 31;
  const int wv   = threadIdx.x >> 5;
  const int lm   = lane & 15;
  const int hi   = lane >> 4;
  const int bh   = blockIdx.y;
  const int bb   = bh >> 4;        // H = 16
  const int hh   = bh & 15;
  const int q0   = blockIdx.x * 64 + wv * 16;

  const _Float16* kbase = Km + (size_t)(bb * kS) * kE + hh * kD;
  const _Float16* vbase = Vt + (size_t)(bh * kD) * kS;
  const unsigned ldsK[2] = {(unsigned)(uintptr_t)&Kbuf[0][0][0],
                            (unsigned)(uintptr_t)&Kbuf[1][0][0]};
  const unsigned ldsV[2] = {(unsigned)(uintptr_t)&Vbuf[0][0][0],
                            (unsigned)(uintptr_t)&Vbuf[1][0][0]};

  // Q rows as B-matrix columns (contiguous over d per lane).
  const _Float16* qp = Qm + (size_t)(bb * kS + q0 + lm) * kE + hh * kD;
  V16H bq0, bq1;
  bq0.u[0] = *(const v4u*)(qp + hi * 16);
  bq0.u[1] = *(const v4u*)(qp + hi * 16 + 8);
  bq1.u[0] = *(const v4u*)(qp + 32 + hi * 16);
  bq1.u[1] = *(const v4u*)(qp + 32 + hi * 16 + 8);

  v8f ot[4] = {{}, {}, {}, {}};
  float mi = -3.0e38f, li = 0.0f;

  // Prologue: stage first K/V tiles into buffer 0.
  if (wv == 0) {
    tdm_load_2d(ldsK[0], kbase, /*d0=*/16, /*rows=*/32, /*stride=*/256, 4, 3);
    tdm_load_2d(ldsV[0], vbase, /*d0=*/8, /*rows=*/64, /*stride=*/512, 3, 3);
  }

  int buf = 0;
  for (int kb = 0; kb < kS; kb += 32) {
    if (wv == 0) {
      if (kb + 32 < kS) {  // prefetch next tiles into the other buffer
        tdm_load_2d(ldsK[buf ^ 1], kbase + (size_t)(kb + 32) * kE, 16, 32, 256, 4, 3);
        tdm_load_2d(ldsV[buf ^ 1], vbase + (kb + 32), 8, 64, 512, 3, 3);
        __builtin_amdgcn_s_wait_tensorcnt((short)2);  // current pair done
      } else {
        __builtin_amdgcn_s_wait_tensorcnt((short)0);
      }
    }
    __syncthreads();  // staged tiles visible to all waves

    const _Float16* Kl = &Kbuf[buf][0][0];
    const _Float16* Vl = &Vbuf[buf][0][0];

    // ---- scoresT tiles for 32 keys (A from LDS) ----
    v8f st[2];
#pragma unroll
    for (int cc = 0; cc < 2; ++cc) {
      const _Float16* kp = Kl + (cc * 16 + lm) * 72;
      V16H a0, a1;
      a0.u[0] = *(const v4u*)(kp + hi * 16);
      a0.u[1] = *(const v4u*)(kp + hi * 16 + 8);
      a1.u[0] = *(const v4u*)(kp + 32 + hi * 16);
      a1.u[1] = *(const v4u*)(kp + 32 + hi * 16 + 8);
      v8f acc = {};
      acc = wmma_f16(a0.v, bq0.v, acc);   // d = 0..31
      acc = wmma_f16(a1.v, bq1.v, acc);   // d = 32..63
      st[cc] = acc;
    }

    // ---- online softmax (per q = lane & 15) ----
    float bmx = -3.0e38f;
#pragma unroll
    for (int cc = 0; cc < 2; ++cc)
#pragma unroll
      for (int j = 0; j < 8; ++j) bmx = fmaxf(bmx, st[cc][j]);
    bmx = fmaxf(bmx, __shfl_xor(bmx, 16, 32));
    const float mnew   = fmaxf(mi, bmx);
    const float scalef = __expf(mi - mnew);
    float bsum = 0.0f;
#pragma unroll
    for (int cc = 0; cc < 2; ++cc) {
      v8h pk;
#pragma unroll
      for (int j = 0; j < 8; ++j) {
        const float p = __expf(st[cc][j] - mnew);
        bsum += p;
        pk[j] = (_Float16)p;
      }
      *(v8h*)&plds[wv][lm][cc * 16 + hi * 8] = pk;
    }
    bsum += __shfl_xor(bsum, 16, 32);
    li = li * scalef + bsum;
    mi = mnew;
#pragma unroll
    for (int t = 0; t < 4; ++t)
#pragma unroll
      for (int j = 0; j < 8; ++j) ot[t][j] *= scalef;

    asm volatile("s_wait_dscnt 0" ::: "memory");

    V16H bp;
    bp.u[0] = *(const v4u*)&plds[wv][lm][hi * 16];
    bp.u[1] = *(const v4u*)&plds[wv][lm][hi * 16 + 8];

    // ---- OT += Vt x PT over 4 d-tiles (A from LDS) ----
#pragma unroll
    for (int t = 0; t < 4; ++t) {
      const _Float16* vp = Vl + (t * 16 + lm) * 40;
      V16H av;
      av.u[0] = *(const v4u*)(vp + hi * 16);
      av.u[1] = *(const v4u*)(vp + hi * 16 + 8);
      ot[t] = wmma_f16(av.v, bp.v, ot[t]);
    }

    __syncthreads();  // all waves done with buf before it is overwritten
    buf ^= 1;
  }

  const float inv = 1.0f / li;
  _Float16* op = Om + (size_t)(bb * kS + q0 + lm) * kE + hh * kD;
#pragma unroll
  for (int t = 0; t < 4; ++t) {
    v8h o8;
#pragma unroll
    for (int j = 0; j < 8; ++j) o8[j] = (_Float16)(ot[t][j] * inv);
    *(v8h*)(op + t * 16 + hi * 8) = o8;
  }
}

// ---------------------------------------------------------------------------
extern "C" void kernel_launch(void* const* d_in, const int* in_sizes, int n_in,
                              void* d_out, int out_size, void* d_ws, size_t ws_size,
                              hipStream_t stream) {
  (void)in_sizes; (void)n_in; (void)out_size; (void)ws_size;

  const float* x  = (const float*)d_in[0];
  const float* Wq = (const float*)d_in[1];
  const float* bq = (const float*)d_in[2];
  const float* Wk = (const float*)d_in[3];
  const float* bk = (const float*)d_in[4];
  const float* Wv = (const float*)d_in[5];
  const float* bv = (const float*)d_in[6];
  const float* Wo = (const float*)d_in[7];
  const float* bo = (const float*)d_in[8];
  float* out = (float*)d_out;

  char* ws = (char*)d_ws;
  const size_t szXE = (size_t)kM * kE * sizeof(_Float16);  // 8 MiB
  const size_t szW  = (size_t)kE * kE * sizeof(_Float16);  // 2 MiB
  _Float16* X16  = (_Float16*)ws; ws += szXE;
  _Float16* Wq16 = (_Float16*)ws; ws += szW;
  _Float16* Wk16 = (_Float16*)ws; ws += szW;
  _Float16* Wv16 = (_Float16*)ws; ws += szW;
  _Float16* Wo16 = (_Float16*)ws; ws += szW;
  _Float16* Q16  = (_Float16*)ws; ws += szXE;
  _Float16* K16  = (_Float16*)ws; ws += szXE;
  _Float16* Vt16 = (_Float16*)ws; ws += szXE;
  _Float16* O16  = (_Float16*)ws; ws += szXE;   // total 48 MiB of d_ws

  auto cvt = [&](const float* s, _Float16* d, size_t n) {
    const int n4 = (int)(n / 4);
    int blocks = (n4 + 255) / 256;
    if (blocks > 2048) blocks = 2048;
    cvt_f32_to_f16<<<blocks, 256, 0, stream>>>((const v4f*)s, (v4h*)d, n4);
  };
  cvt(x,  X16,  (size_t)kM * kE);
  cvt(Wq, Wq16, (size_t)kE * kE);
  cvt(Wk, Wk16, (size_t)kE * kE);
  cvt(Wv, Wv16, (size_t)kE * kE);
  cvt(Wo, Wo16, (size_t)kE * kE);

  const dim3 gg(kM / 64, kE / 64);  // 64 x 16 blocks, 4 waves each
  gemm_kernel<0><<<gg, 128, 0, stream>>>(X16, Wq16, bq, Q16, nullptr);
  gemm_kernel<1><<<gg, 128, 0, stream>>>(X16, Wk16, bk, K16, nullptr);
  gemm_kernel<2><<<gg, 128, 0, stream>>>(X16, Wv16, bv, Vt16, nullptr);

  flash_kernel<<<dim3(kS / 64, kB * kH), 128, 0, stream>>>(Q16, K16, Vt16, O16);

  gemm_kernel<3><<<gg, 128, 0, stream>>>(O16, Wo16, bo, nullptr, out);
}
